// SeqVLAD_18854906430119
// MI455X (gfx1250) — matrix operand order repaired and checked by reading, not compiled, not run on hardware
//
#include <hip/hip_runtime.h>
#include <hip/hip_bf16.h>

typedef float        v8f   __attribute__((ext_vector_type(8)));
typedef __bf16       v16bf __attribute__((ext_vector_type(16)));
typedef __bf16       bf16_t;
typedef unsigned int u32x4 __attribute__((ext_vector_type(4)));
typedef unsigned int u32x8 __attribute__((ext_vector_type(8)));

union V16BF { v16bf v; bf16_t e[16]; };
union V8F   { v8f   v; float  e[8];  };

#define B_     32
#define S_     5
#define D_     128
#define K_     64
#define HW_    1200      /* pixels per (n,s) slice */
#define PT     128       /* pixel tile */
#define PTpad  136
#define NIT    10        /* ceil(1200/128) tiles per slice */

// ---------------------------------------------------------------------------
// TDM: async 2D tile load [D_ rows x PT cols, f32, row stride HW_] -> LDS.
// Descriptor per CDNA5 ISA 8.3/8.4 (group0: count/lds_addr/global_addr/type,
// group1: data_size/dims/tile/strides). Issued via inline asm (portable
// across ROCm-7.2 / amdgpu-toolchain builtin-arity split). Tracked: TENSORcnt.
// ---------------------------------------------------------------------------
__device__ __forceinline__ void tdm_load_2d(const float* gaddr,
                                            unsigned lds_off,
                                            unsigned rem_pix) {
  unsigned long long ga = (unsigned long long)(size_t)gaddr;
  u32x4 g0;
  g0[0] = 1u;                                            // count=1, user desc
  g0[1] = lds_off;                                       // lds_addr [63:32]
  g0[2] = (unsigned)ga;                                  // global_addr lo
  g0[3] = (unsigned)((ga >> 32) & 0x1FFFFFFull)          // global_addr [120:96]
        | (2u << 30);                                    // type=2 ("image")
  u32x8 g1;
  g1[0] = 2u << 16;                                      // data_size=4B, no mask
  g1[1] = (rem_pix & 0xFFFFu) << 16;                     // tensor_dim0 lo16
  g1[2] = (rem_pix >> 16) | ((unsigned)D_ << 16);        // dim0 hi | tensor_dim1 lo
  g1[3] = ((unsigned)PT << 16);                          // dim1 hi=0 | tile_dim0
  g1[4] = (unsigned)D_;                                  // tile_dim1 (tile_dim2=0)
  g1[5] = (unsigned)HW_;                                 // tensor_dim0_stride lo32
  g1[6] = 0u;                                            // stride hi / dim1_stride
  g1[7] = 0u;
  asm volatile("tensor_load_to_lds %0, %1" :: "s"(g0), "s"(g1) : "memory");
}

// ---------------------------------------------------------------------------
// Kernel 1: TDM-streamed normalize + GEMM1(WMMA) + softmax + GEMM2(WMMA)
// One workgroup (8 wave32) per (n,s) slice of 1200 pixels.
// ---------------------------------------------------------------------------
__global__ __launch_bounds__(256) void vlad_acc_kernel(
    const float* __restrict__ x, const float* __restrict__ Wt,
    float* __restrict__ vacc, float* __restrict__ asum)
{
  __shared__ float  raw[2][D_ * PT];    // TDM double buffer    (128 KB)
  __shared__ bf16_t Wl[K_ * D_];        // conv weights, bf16   (16 KB)
  __shared__ bf16_t xn[D_ * PTpad];     // normalized x tile    (34 KB)
  __shared__ bf16_t al[K_ * PTpad];     // soft-assign tile     (17 KB)
  __shared__ float  lg[K_ * PT];        // logits               (32 KB)
  __shared__ float  ssq[2 * PT];
  __shared__ float  rno[PT];

  const int tid  = threadIdx.x;
  const int lane = tid & 31;
  const int wv   = tid >> 5;            // wave id 0..7
  const int n    = blockIdx.x / S_;
  const int s    = blockIdx.x % S_;
  const float* slice = x + (size_t)(n * S_ + s) * D_ * HW_;

  // preload W (fp32 -> bf16) into LDS
  for (int i = tid; i < K_ * D_; i += 256) Wl[i] = (bf16_t)Wt[i];

  // prologue: async TDM load of tile 0
  if (wv == 0)
    tdm_load_2d(slice, (unsigned)(size_t)&raw[0][0], (unsigned)HW_);

  V8F acc2[4];
  #pragma unroll
  for (int j = 0; j < 4; ++j)
    #pragma unroll
    for (int e = 0; e < 8; ++e) acc2[j].e[e] = 0.0f;
  float asum_loc = 0.0f;

  const int lm  = lane & 15;            // M / N position within a 16-wide tile
  const int kh  = lane >> 4;            // lane half (ISA A/B/C layout)
  const int mt  = wv >> 1;              // this wave's M(=k) tile: 0..3
  const int ntb = (wv & 1) * 4;         // this wave's N tile base

  int cur = 0;
  __syncthreads();

  for (int it = 0; it < NIT; ++it) {
    if (wv == 0) __builtin_amdgcn_s_wait_tensorcnt(0);  // raw[cur] landed
    __syncthreads();

    // ---- Phase 1: per-pixel L2 normalize over D (from TDM-staged tile) ----
    {
      const int pi = tid & 127, hf = tid >> 7;   // 2 threads/pixel, 64 ch each
      const bool valid = (it * PT + pi) < HW_;
      const float* rb = &raw[cur][0];
      float ss = 0.0f;
      #pragma unroll 4
      for (int cc = 0; cc < 64; ++cc) {
        float v = valid ? rb[(hf * 64 + cc) * PT + pi] : 0.0f;
        ss += v * v;
        xn[(hf * 64 + cc) * PTpad + pi] = (bf16_t)v;
      }
      ssq[hf * PT + pi] = ss;
    }
    __syncthreads();
    if (tid < PT) {
      float t = ssq[tid] + ssq[PT + tid];
      rno[tid] = 1.0f / fmaxf(sqrtf(t), 1e-12f);
    }
    __syncthreads();
    {
      const int pi = tid & 127, hf = tid >> 7;
      const float rn = rno[pi];
      #pragma unroll 4
      for (int cc = 0; cc < 64; ++cc) {
        const int idx = (hf * 64 + cc) * PTpad + pi;
        xn[idx] = (bf16_t)((float)xn[idx] * rn);
      }
    }
    __syncthreads();   // xn ready; raw[cur] fully consumed

    // kick off the next tile's TDM transfer: overlaps all the math below
    if (wv == 0 && (it + 1) < NIT)
      tdm_load_2d(slice + (size_t)(it + 1) * PT,
                  (unsigned)(size_t)&raw[cur ^ 1][0],
                  (unsigned)(HW_ - (it + 1) * PT));

    // ---- GEMM1: logits[K,PT] = W[K,D] x xn[D,PT] via v_wmma bf16 ----
    for (int j = 0; j < 4; ++j) {
      const int nt = ntb + j;             // pixel tile 0..7
      V8F c;
      #pragma unroll
      for (int e = 0; e < 8; ++e) c.e[e] = 0.0f;
      for (int kk = 0; kk < 4; ++kk) {    // d in blocks of 32
        V16BF a, b;
        #pragma unroll
        for (int v = 0; v < 8; ++v) {
          const int g = v >> 2, kp = v & 3;
          const int kOff = g * 16 + kh * 8 + kp * 2;    // ISA 16-bit A layout
          const int d = kk * 32 + kOff;
          a.e[2 * v]     = Wl[(mt * 16 + lm) * D_ + d];
          a.e[2 * v + 1] = Wl[(mt * 16 + lm) * D_ + d + 1];
          const int dr = kk * 32 + (v >> 2) * 16 + kh * 8 + (v & 3) * 2;     // ISA 16-bit B layout
          b.e[2 * v]     = xn[dr * PTpad + nt * 16 + lm];
          b.e[2 * v + 1] = xn[(dr + 1) * PTpad + nt * 16 + lm];
        }
        c.v = __builtin_amdgcn_wmma_f32_16x16x32_bf16(
            false, a.v, false, b.v, (short)0, c.v, false, false);
      }
      #pragma unroll
      for (int v = 0; v < 8; ++v)         // C layout: lane=N, VGPR=M (+8 hi half)
        lg[(mt * 16 + kh * 8 + v) * PT + nt * 16 + lm] = c.e[v];
    }
    __syncthreads();

    // ---- Softmax over K per pixel (exact f32) ----
    if (tid < PT) {
      const bool valid = (it * PT + tid) < HW_;
      float m = -3.4e38f;
      for (int k = 0; k < K_; ++k) m = fmaxf(m, lg[k * PT + tid]);
      float sum = 0.0f;
      for (int k = 0; k < K_; ++k) sum += __expf(lg[k * PT + tid] - m);
      const float rs = valid ? (1.0f / sum) : 0.0f;
      for (int k = 0; k < K_; ++k)
        al[k * PTpad + tid] = (bf16_t)(__expf(lg[k * PT + tid] - m) * rs);
    }
    __syncthreads();

    // per-k sum of soft assignments (for centroid term)
    if (tid < K_) {
      float sm = 0.0f;
      for (int p = 0; p < PT; ++p) sm += (float)al[tid * PTpad + p];
      asum_loc += sm;
    }

    // ---- GEMM2: vlad[K,D] += a[K,PT] x xn^T[PT,D] via v_wmma bf16 ----
    for (int pp = 0; pp < 4; ++pp) {      // inner p in blocks of 32
      V16BF a2;
      #pragma unroll
      for (int v = 0; v < 8; ++v) {
        const int g = v >> 2, kp = v & 3;
        const int kOff = g * 16 + kh * 8 + kp * 2;
        const int pr = pp * 32 + kOff;
        a2.e[2 * v]     = al[(mt * 16 + lm) * PTpad + pr];
        a2.e[2 * v + 1] = al[(mt * 16 + lm) * PTpad + pr + 1];
      }
      for (int j = 0; j < 4; ++j) {
        const int nt = ntb + j;           // d tile 0..7
        V16BF b2;
        #pragma unroll
        for (int v = 0; v < 8; ++v) {
          const int pr = pp * 32 + (v >> 2) * 16 + kh * 8 + (v & 3) * 2;
          b2.e[2 * v]     = xn[(nt * 16 + lm) * PTpad + pr];
          b2.e[2 * v + 1] = xn[(nt * 16 + lm) * PTpad + pr + 1];
        }
        acc2[j].v = __builtin_amdgcn_wmma_f32_16x16x32_bf16(
            false, a2.v, false, b2.v, (short)0, acc2[j].v, false, false);
      }
    }
    __syncthreads();
    cur ^= 1;
  }

  // ---- write partial vlad / asum to workspace ----
  for (int j = 0; j < 4; ++j) {
    const int nt = ntb + j;
    #pragma unroll
    for (int v = 0; v < 8; ++v) {
      const int k = mt * 16 + kh * 8 + v;
      const int d = nt * 16 + lm;
      atomicAdd(&vacc[((size_t)n * K_ + k) * D_ + d], acc2[j].e[v]);
    }
  }
  if (tid < K_) atomicAdd(&asum[n * K_ + tid], asum_loc);
}

// ---------------------------------------------------------------------------
// Kernel 2: centroid subtraction, intra-norm, final L2 norm
// ---------------------------------------------------------------------------
__global__ __launch_bounds__(256) void vlad_finalize_kernel(
    const float* __restrict__ vacc, const float* __restrict__ asum,
    const float* __restrict__ cent, float* __restrict__ out)
{
  __shared__ float v[K_ * D_];
  __shared__ float rn[K_];
  __shared__ float red[256];
  __shared__ float tot;
  const int n = blockIdx.x;
  const int tid = threadIdx.x;

  for (int idx = tid; idx < K_ * D_; idx += 256) {
    const int k = idx >> 7;
    v[idx] = vacc[(size_t)n * K_ * D_ + idx] - cent[idx] * asum[n * K_ + k];
  }
  __syncthreads();
  if (tid < K_) {
    float ss = 0.0f;
    for (int d = 0; d < D_; ++d) { float t = v[tid * D_ + d]; ss += t * t; }
    rn[tid] = 1.0f / fmaxf(sqrtf(ss), 1e-12f);
  }
  __syncthreads();
  float loc = 0.0f;
  for (int idx = tid; idx < K_ * D_; idx += 256) {
    float t = v[idx] * rn[idx >> 7];
    v[idx] = t;
    loc += t * t;
  }
  red[tid] = loc;
  __syncthreads();
  if (tid == 0) {
    float sgl = 0.0f;
    for (int i = 0; i < 256; ++i) sgl += red[i];
    tot = 1.0f / fmaxf(sqrtf(sgl), 1e-12f);
  }
  __syncthreads();
  const float r2 = tot;
  for (int idx = tid; idx < K_ * D_; idx += 256)
    out[(size_t)n * K_ * D_ + idx] = v[idx] * r2;
}

// ---------------------------------------------------------------------------
extern "C" void kernel_launch(void* const* d_in, const int* in_sizes, int n_in,
                              void* d_out, int out_size, void* d_ws, size_t ws_size,
                              hipStream_t stream) {
  const float* x = (const float*)d_in[0];
  const float* w = (const float*)d_in[1];
  const float* c = (const float*)d_in[2];
  float* out  = (float*)d_out;
  float* vacc = (float*)d_ws;                         // 32*64*128 f32
  float* asum = vacc + (size_t)B_ * K_ * D_;          // 32*64 f32

  const size_t zbytes = ((size_t)B_ * K_ * D_ + (size_t)B_ * K_) * sizeof(float);
  hipMemsetAsync(d_ws, 0, zbytes, stream);

  vlad_acc_kernel<<<dim3(B_ * S_), dim3(256), 0, stream>>>(x, w, vacc, asum);
  vlad_finalize_kernel<<<dim3(B_), dim3(256), 0, stream>>>(vacc, asum, c, out);
}
